// CustomCausalAttention_23261542875729
// MI455X (gfx1250) — matrix operand order repaired and checked by reading, not compiled, not run on hardware
//
#include <hip/hip_runtime.h>

typedef __attribute__((ext_vector_type(16))) __bf16 v16bf;
typedef __attribute__((ext_vector_type(8)))  float  v8f;
typedef __attribute__((ext_vector_type(4)))  unsigned int u32x4;
typedef __attribute__((ext_vector_type(4)))  float  f32x4;

#define WAVES    8
#define BLOCK_M  (16 * WAVES)
#define SM_SCALE 0.125f

constexpr int Bb = 2, Hh = 16, Ss = 2048, Dd = 64;
constexpr int KSTR = 72;   // bf16 stride, K tile row  (64 data + 8 pad)
constexpr int VSTR = 40;   // bf16 stride, V^T tile row (32 data + 8 pad)
constexpr int PSTR = 40;   // bf16 stride, P tile row   (32 data + 8 pad)

union FragBF { v16bf v; u32x4 q[2]; };

// CDNA5 async copy: global -> LDS, 128b per lane, tracked by ASYNCcnt.
__device__ __forceinline__ void async_ld_b128(const void* g, const void* l) {
    unsigned lds = (unsigned)(uintptr_t)l;                 // LDS offset (addr[31:0])
    unsigned long long ga = (unsigned long long)(uintptr_t)g;
    asm volatile("global_load_async_to_lds_b128 %0, %1, off"
                 :: "v"(lds), "v"(ga) : "memory");
}

__global__ __launch_bounds__(256, 1)
void rbf_causal_attn_wmma(const float* __restrict__ q,
                          const float* __restrict__ kmat,
                          const float* __restrict__ vmat,
                          float* __restrict__ o)
{
    __shared__ alignas(16) float  rawk[2][32 * 64];   // double-buffered fp32 K tile
    __shared__ alignas(16) float  rawv[2][32 * 64];   // double-buffered fp32 V tile
    __shared__ alignas(16) __bf16 kt[32 * KSTR];      // bf16 K tile, row-major [key][d]
    __shared__ alignas(16) __bf16 vt[64 * VSTR];      // bf16 V tile, transposed [d][key]
    __shared__ alignas(16) __bf16 pt[WAVES * 16 * PSTR];
    __shared__ alignas(16) float  ksqs[32];           // scaled ||k||^2 per tile key

    const int tid  = threadIdx.x;
    const int lane = tid & 31;
    const int wave = tid >> 5;
    const int nl   = lane & 15;
    const int half = lane >> 4;
    const int bh   = blockIdx.x;
    const int m_tile = blockIdx.y * BLOCK_M + wave * 16;
    const int n_end  = blockIdx.y * BLOCK_M + BLOCK_M;   // block-uniform causal bound

    const float* qb = q    + (size_t)bh * Ss * Dd;
    const float* kb = kmat + (size_t)bh * Ss * Dd;
    const float* vb = vmat + (size_t)bh * Ss * Dd;
    float*       ob = o    + (size_t)bh * Ss * Dd;

    // ---- prologue: Q strip as two bf16 A-fragments + scaled row ||q||^2
    v16bf qa[2];
    float qss = 0.f;
    const float* qrowp = qb + (size_t)(m_tile + nl) * Dd;
#pragma unroll
    for (int f = 0; f < 2; ++f)
#pragma unroll
        for (int i = 0; i < 8; ++i) {
            int kd = 32 * f + ((i < 4) ? 0 : 16) + 8 * half + 2 * (i & 3);
            float x0 = qrowp[kd], x1 = qrowp[kd + 1];
            qss += x0 * x0 + x1 * x1;
            qa[f][2 * i]     = (__bf16)x0;
            qa[f][2 * i + 1] = (__bf16)x1;
        }
    qss = SM_SCALE * (qss + __shfl_xor(qss, 16, 32));
    float qrow[8];
#pragma unroll
    for (int r = 0; r < 8; ++r) qrow[r] = __shfl(qss, r + 8 * half, 32);

    v8f acc[4] = {};
    __bf16* myp = pt + wave * 16 * PSTR;

    // ---- prefetch tile 0 (4 async b128 per thread: K tile + V tile, 8KB each)
#pragma unroll
    for (int p = 0; p < 2; ++p) {
        int idx = tid + p * 256;
        async_ld_b128(kb + idx * 4, &rawk[0][idx * 4]);
        async_ld_b128(vb + idx * 4, &rawv[0][idx * 4]);
    }

    for (int kb0 = 0, it = 0; kb0 < n_end; kb0 += 32, ++it) {
        const int buf = it & 1;

        asm volatile("s_wait_asynccnt 0x0" ::: "memory");   // tile `buf` landed
        __syncthreads();

        // issue async prefetch of next tile into other buffer (overlaps below)
        if (kb0 + 32 < n_end) {
            const float* kg = kb + (size_t)(kb0 + 32) * Dd;
            const float* vg = vb + (size_t)(kb0 + 32) * Dd;
#pragma unroll
            for (int p = 0; p < 2; ++p) {
                int idx = tid + p * 256;
                async_ld_b128(kg + idx * 4, &rawk[buf ^ 1][idx * 4]);
                async_ld_b128(vg + idx * 4, &rawv[buf ^ 1][idx * 4]);
            }
        }

        // ---- cooperative fp32 -> bf16 conversion
        {
            // K: thread owns one key's 8-dim chunk; shfl-reduce ||k||^2 across the 8 owners
            const int e = tid * 8, key = e >> 6, d = e & 63;
            f32x4 a = *(const f32x4*)&rawk[buf][e];
            f32x4 b = *(const f32x4*)&rawk[buf][e + 4];
            float ps = a.x*a.x + a.y*a.y + a.z*a.z + a.w*a.w +
                       b.x*b.x + b.y*b.y + b.z*b.z + b.w*b.w;
            ps += __shfl_xor(ps, 1, 32);
            ps += __shfl_xor(ps, 2, 32);
            ps += __shfl_xor(ps, 4, 32);
            if ((tid & 7) == 0) ksqs[tid >> 3] = SM_SCALE * ps;

            union { __bf16 h[8]; u32x4 qv; } pk;
            pk.h[0]=(__bf16)a.x; pk.h[1]=(__bf16)a.y; pk.h[2]=(__bf16)a.z; pk.h[3]=(__bf16)a.w;
            pk.h[4]=(__bf16)b.x; pk.h[5]=(__bf16)b.y; pk.h[6]=(__bf16)b.z; pk.h[7]=(__bf16)b.w;
            *(u32x4*)&kt[key * KSTR + d] = pk.qv;

            // V: thread owns a key-pair x 4-dim patch; packed b32 transposed stores
            const int kp = tid >> 4, dc = (tid & 15) * 4;
            f32x4 v0 = *(const f32x4*)&rawv[buf][(2 * kp)     * 64 + dc];
            f32x4 v1 = *(const f32x4*)&rawv[buf][(2 * kp + 1) * 64 + dc];
            union { __bf16 h[2]; unsigned u; } w;
            w.h[0]=(__bf16)v0.x; w.h[1]=(__bf16)v1.x; *(unsigned*)&vt[(dc+0)*VSTR + 2*kp] = w.u;
            w.h[0]=(__bf16)v0.y; w.h[1]=(__bf16)v1.y; *(unsigned*)&vt[(dc+1)*VSTR + 2*kp] = w.u;
            w.h[0]=(__bf16)v0.z; w.h[1]=(__bf16)v1.z; *(unsigned*)&vt[(dc+2)*VSTR + 2*kp] = w.u;
            w.h[0]=(__bf16)v0.w; w.h[1]=(__bf16)v1.w; *(unsigned*)&vt[(dc+3)*VSTR + 2*kp] = w.u;
        }
        __syncthreads();

        // ---- per-wave compute; waves past their diagonal skip (no barrier inside)
        if (kb0 < m_tile + 16) {
            // hoist all K-fragment loads, then 4 back-to-back WMMAs
            FragBF kf[2][2];
#pragma unroll
            for (int s = 0; s < 2; ++s) {
                const __bf16* kr = kt + (16 * s + nl) * KSTR + 8 * half;
                kf[s][0].q[0] = *(const u32x4*)(kr);
                kf[s][0].q[1] = *(const u32x4*)(kr + 16);
                kf[s][1].q[0] = *(const u32x4*)(kr + 32);
                kf[s][1].q[1] = *(const u32x4*)(kr + 48);
            }
            float ksq[2] = { ksqs[nl], ksqs[16 + nl] };

            v8f sc[2] = {};
            sc[0] = __builtin_amdgcn_wmma_f32_16x16x32_bf16(false, qa[0], false, kf[0][0].v,
                                                            (short)0, sc[0], false, false);
            sc[0] = __builtin_amdgcn_wmma_f32_16x16x32_bf16(false, qa[1], false, kf[0][1].v,
                                                            (short)0, sc[0], false, false);
            sc[1] = __builtin_amdgcn_wmma_f32_16x16x32_bf16(false, qa[0], false, kf[1][0].v,
                                                            (short)0, sc[1], false, false);
            sc[1] = __builtin_amdgcn_wmma_f32_16x16x32_bf16(false, qa[1], false, kf[1][1].v,
                                                            (short)0, sc[1], false, false);

            // start V-fragment loads now; they fly during the exp section
            FragBF vf[4];
#pragma unroll
            for (int c4 = 0; c4 < 4; ++c4) {
                const __bf16* vr = vt + (16 * c4 + nl) * VSTR + 8 * half;
                vf[c4].q[0] = *(const u32x4*)(vr);
                vf[c4].q[1] = *(const u32x4*)(vr + 16);
            }

            // logits -> exp -> causal mask -> bf16 P tile (C layout -> row major)
#pragma unroll
            for (int s = 0; s < 2; ++s)
#pragma unroll
                for (int r = 0; r < 8; ++r) {
                    int  mg = m_tile + r + 8 * half;
                    int  ng = kb0 + 16 * s + nl;
                    float lg = 2.f * SM_SCALE * sc[s][r] - qrow[r] - ksq[s];
                    float p  = (mg >= ng) ? __expf(lg) : 0.f;
                    myp[(r + 8 * half) * PSTR + 16 * s + nl] = (__bf16)p;
                }

            __builtin_amdgcn_wave_barrier();
            asm volatile("s_wait_dscnt 0x0" ::: "memory");   // intra-wave only

            FragBF pa;
            pa.q[0] = *(const u32x4*)&myp[nl * PSTR + 8 * half];
            pa.q[1] = *(const u32x4*)&myp[nl * PSTR + 16 + 8 * half];
            __builtin_amdgcn_wave_barrier();

            // P (16x32) @ V (32x64): 4 back-to-back WMMAs, fragments already resident
            acc[0] = __builtin_amdgcn_wmma_f32_16x16x32_bf16(false, pa.v, false, vf[0].v,
                                                             (short)0, acc[0], false, false);
            acc[1] = __builtin_amdgcn_wmma_f32_16x16x32_bf16(false, pa.v, false, vf[1].v,
                                                             (short)0, acc[1], false, false);
            acc[2] = __builtin_amdgcn_wmma_f32_16x16x32_bf16(false, pa.v, false, vf[2].v,
                                                             (short)0, acc[2], false, false);
            acc[3] = __builtin_amdgcn_wmma_f32_16x16x32_bf16(false, pa.v, false, vf[3].v,
                                                             (short)0, acc[3], false, false);
        }
    }

    // ---- store fp32 output tile (C layout: lane = col, VGPR r = rows r, r+8)
#pragma unroll
    for (int c4 = 0; c4 < 4; ++c4)
#pragma unroll
        for (int r = 0; r < 8; ++r)
            ob[(size_t)(m_tile + r + 8 * half) * Dd + 16 * c4 + nl] = acc[c4][r];
}

extern "C" void kernel_launch(void* const* d_in, const int* in_sizes, int n_in,
                              void* d_out, int out_size, void* d_ws, size_t ws_size,
                              hipStream_t stream) {
    (void)in_sizes; (void)n_in; (void)out_size; (void)d_ws; (void)ws_size;
    const float* q = (const float*)d_in[0];
    const float* k = (const float*)d_in[1];
    const float* v = (const float*)d_in[2];
    float* out = (float*)d_out;

    dim3 grid(Bb * Hh, Ss / BLOCK_M);
    rbf_causal_attn_wmma<<<grid, 32 * WAVES, 0, stream>>>(q, k, v, out);
}
